// Cost_17411797418344
// MI455X (gfx1250) — matrix operand order repaired and checked by reading, not compiled, not run on hardware
//
#include <hip/hip_runtime.h>

// MI455X / gfx1250 cost-volume (correlation) kernel.
// cost[b,d,h,w] = sum_c f1[b,c,h,w] * f2[b,c,h+dh,w+dw], (dh,dw) in [-4,4]^2.
//
// Per (b, h, 16-wide w-tile) one wave32 computes, for each dh, the 16x32 band
// R[m,n] = sum_c f1[c,w0+m] * f2[c,w0-4+n] with v_wmma_f32_16x16x32_bf16,
// using a bf16 hi/lo split of the fp32 inputs (3 products -> ~fp32 accuracy).
// cost[m,dw] = R[m, m+dw] extracted via LDS for coalesced stores.
//
// Round-1 fix: boundary masking on B loads now uses clamped (always-valid)
// addresses + value select instead of conditional loads, eliminating the
// per-load s_and_saveexec/s_cbranch_execz diamonds seen in the round-0 asm.

typedef __attribute__((ext_vector_type(16))) __bf16 v16bf;
typedef __attribute__((ext_vector_type(8)))  float  v8f;

#define BB 8
#define CC 96
#define HH 160
#define WW 320
#define MOFF 4          // max offset (both dims)
#define ND 9            // offsets per dim
#define NDD 81          // total offsets

__device__ __forceinline__ void split_bf16(float x, __bf16& hi, __bf16& lo) {
    hi = (__bf16)x;
    lo = (__bf16)(x - (float)hi);
}

__global__ __launch_bounds__(32)
void cost_volume_wmma(const float* __restrict__ f1,
                      const float* __restrict__ f2,
                      float* __restrict__ out) {
    const int lane = threadIdx.x;        // 0..31
    const int half = lane >> 4;          // 0 or 1 (lane group)
    const int lm   = lane & 15;          // row m (for A/C) or col n (for B)
    const int w0   = blockIdx.x * 16;    // w tile base, 20 tiles
    const int h    = blockIdx.y;         // 0..159
    const int b    = blockIdx.z;         // 0..7
    const size_t HW = (size_t)HH * WW;

    // R band: cols 0..15 from N-tile 0 (w' = w0-4 .. w0+11), 16..31 from N-tile 1.
    __shared__ float R[16][36];          // +4 pad vs 32 to dodge bank conflicts

    // Per-lane B column for each N-tile: clamp into [0, W-1] so loads are
    // always in-bounds (unconditional), and remember a 0/1 mask for the value.
    int   wp_c[2];
    float wmask[2];
    #pragma unroll
    for (int nt = 0; nt < 2; ++nt) {
        const int wp = w0 - MOFF + nt * 16 + lm;
        const bool inw = (wp >= 0) && (wp < WW);
        wp_c[nt]  = inw ? wp : (wp < 0 ? 0 : WW - 1);
        wmask[nt] = inw ? 1.0f : 0.0f;
    }

    // ---- Load A = f1[b, :, h, w0..w0+15], split into bf16 hi/lo, keep resident ----
    // 16-bit A 16x32 layout (ISA 7.12.2): lane(half,m); VGPR v holds
    // K = 2*(v&3) + 16*(v>>2) + 8*half, and K+1 (packed lo/hi 16 bits).
    v16bf Ah[3], Al[3];
    const float* f1p = f1 + (size_t)b * CC * HW + (size_t)h * WW + (w0 + lm);
    #pragma unroll
    for (int kb = 0; kb < 3; ++kb) {
        v16bf ah, al;
        #pragma unroll
        for (int v = 0; v < 8; ++v) {
            const int kbase = ((v & 3) * 2) + ((v >> 2) * 16) + half * 8;
            #pragma unroll
            for (int e = 0; e < 2; ++e) {
                const int c = kb * 32 + kbase + e;
                float x = f1p[(size_t)c * HW];
                __bf16 hi, lo; split_bf16(x, hi, lo);
                ah[2 * v + e] = hi;
                al[2 * v + e] = lo;
            }
        }
        Ah[kb] = ah; Al[kb] = al;
    }

    for (int dh = 0; dh < ND; ++dh) {
        const int h2 = h + dh - MOFF;
        const size_t obase =
            (((size_t)b * NDD + (size_t)dh * ND) * HH + h) * WW + w0;

        if (h2 < 0 || h2 >= HH) {        // block-uniform: zero-pad rows
            #pragma unroll
            for (int it = 0; it < 5; ++it) {
                const int idx = it * 32 + lane;       // dw*16 + m
                if (idx < ND * 16)
                    out[obase + (size_t)(idx >> 4) * HW + (idx & 15)] = 0.0f;
            }
            continue;
        }

        const float* f2row = f2 + (size_t)b * CC * HW + (size_t)h2 * WW;
        // Prefetch next dh's feat2 row (speculative; dropped if invalid).
        if (h2 + 1 < HH)
            __builtin_prefetch(f2row + WW + w0 + lm, 0, 0);

        #pragma unroll
        for (int nt = 0; nt < 2; ++nt) {
            const float* bptr = f2row + wp_c[nt];     // always-valid column base
            const float  msk  = wmask[nt];
            v8f acc = {};
            #pragma unroll
            for (int kb = 0; kb < 3; ++kb) {
                // 16-bit B 32x16 layout: lane(half,n); VGPR v holds
                // K = 2*(v&3) + 16*(v>>2) + 8*half, and K+1.
                v16bf bh, bl;
                #pragma unroll
                for (int v = 0; v < 8; ++v) {
                    #pragma unroll
                    for (int e = 0; e < 2; ++e) {
                        const int c = kb * 32 + ((v & 3) * 2) + ((v >> 2) * 16) + half * 8 + e;
                        float x = bptr[(size_t)c * HW] * msk;   // unconditional load
                        __bf16 hi, lo; split_bf16(x, hi, lo);
                        bh[2 * v + e] = hi;
                        bl[2 * v + e] = lo;
                    }
                }
                // fp32-accurate product via 3 bf16 WMMAs: AhBh + AhBl + AlBh
                acc = __builtin_amdgcn_wmma_f32_16x16x32_bf16(
                          false, Ah[kb], false, bh, (short)0, acc, false, false);
                acc = __builtin_amdgcn_wmma_f32_16x16x32_bf16(
                          false, Ah[kb], false, bl, (short)0, acc, false, false);
                acc = __builtin_amdgcn_wmma_f32_16x16x32_bf16(
                          false, Al[kb], false, bh, (short)0, acc, false, false);
            }
            // C/D f32 layout: VGPR g -> row g + 8*half, col = lane&15.
            #pragma unroll
            for (int g = 0; g < 8; ++g)
                R[g + 8 * half][nt * 16 + lm] = acc[g];
        }
        __syncthreads();

        // cost[m, dw] = R[m, m + dw]  (w' = w0 + m + (dw-4) -> col m+dw in band)
        #pragma unroll
        for (int it = 0; it < 5; ++it) {
            const int idx = it * 32 + lane;           // dw*16 + m, coalesced in m
            if (idx < ND * 16) {
                const int dw = idx >> 4, m = idx & 15;
                out[obase + (size_t)dw * HW + m] = R[m][m + dw];
            }
        }
        __syncthreads();
    }
}

extern "C" void kernel_launch(void* const* d_in, const int* in_sizes, int n_in,
                              void* d_out, int out_size, void* d_ws, size_t ws_size,
                              hipStream_t stream) {
    (void)in_sizes; (void)n_in; (void)out_size; (void)d_ws; (void)ws_size;
    const float* f1 = (const float*)d_in[0];
    const float* f2 = (const float*)d_in[1];
    // d_in[2], d_in[3] are max_offset_h/w == 4, baked in at compile time.
    float* out = (float*)d_out;
    dim3 grid(WW / 16, HH, BB);   // (20, 160, 8) blocks, one wave32 each
    cost_volume_wmma<<<grid, dim3(32), 0, stream>>>(f1, f2, out);
}
